// ChannelTransformerStage_2714419331295
// MI455X (gfx1250) — matrix operand order repaired and checked by reading, not compiled, not run on hardware
//
#include <hip/hip_runtime.h>
#include <hip/hip_bf16.h>

typedef __bf16 bf16_t;
typedef __attribute__((ext_vector_type(16))) __bf16 v16bf;
typedef __attribute__((ext_vector_type(8)))  __bf16 v8bf;
typedef __attribute__((ext_vector_type(8)))  float  v8f;

#define H_   4
#define C1_  64
#define C2_  128
#define KV_  192
#define MLP_ 256
#define B_   8
#define N_   3136
#define BN_  (B_ * N_)

__device__ inline float wave_sum(float v) {
#pragma unroll
  for (int m = 16; m >= 1; m >>= 1) v += __shfl_xor(v, m, 32);
  return v;
}
__device__ inline float wave_max(float v) {
#pragma unroll
  for (int m = 16; m >= 1; m >>= 1) v = fmaxf(v, __shfl_xor(v, m, 32));
  return v;
}

// K-index within a 32-wide chunk for the 16x32 bf16 A fragment (ISA 7.12.2):
// lanes 0-15:  e<8 -> K=e,   e>=8 -> K=e+8   (0..7, 16..23)
// lanes 16-31: e<8 -> K=e+8, e>=8 -> K=e+16  (8..15, 24..31)
__device__ inline int a_kidx(int e, int hi) {
  return e + (e < 8 ? (hi ? 8 : 0) : (hi ? 16 : 8));
}

// ---------------------------------------------------------------------------
// CDNA5 async global->LDS copy (16B per lane), tracked by ASYNCcnt.
// ---------------------------------------------------------------------------
__device__ __forceinline__ void async_copy16(void* lds, const void* g) {
  unsigned l = (unsigned)(unsigned long long)lds;           // LDS byte address
  unsigned long long ga = (unsigned long long)g;
  asm volatile("global_load_async_to_lds_b128 %0, %1, off"
               :: "v"(l), "v"(ga) : "memory");
}
__device__ __forceinline__ void async_wait0() {
  asm volatile("s_wait_asynccnt 0x0" ::: "memory");
}

// ---------------------------------------------------------------------------
// f32 -> bf16 convert
// ---------------------------------------------------------------------------
__global__ void f32_to_bf16_kernel(const float* __restrict__ src,
                                   bf16_t* __restrict__ dst, int n) {
  int i = blockIdx.x * blockDim.x + threadIdx.x;
  if (i < n) dst[i] = (bf16_t)src[i];
}

// ---------------------------------------------------------------------------
// Fused: cx1 = LN(emb1) (C1), ea = LN(concat(emb1,emb2)) (KV). 1 wave per row.
// ---------------------------------------------------------------------------
__global__ __launch_bounds__(256) void ln_concat_kernel(
    const float* __restrict__ emb1, const float* __restrict__ emb2,
    const float* __restrict__ g1, const float* __restrict__ b1,
    const float* __restrict__ ga, const float* __restrict__ ba,
    bf16_t* __restrict__ cx1, bf16_t* __restrict__ ea) {
  const int row  = blockIdx.x * 8 + (threadIdx.x >> 5);
  const int lane = threadIdx.x & 31;
  const float* r1 = emb1 + (long)row * C1_;
  const float* r2 = emb2 + (long)row * C2_;

  float x0 = r1[lane], x1 = r1[lane + 32];
  float m  = wave_sum(x0 + x1) * (1.0f / 64.0f);
  float d0 = x0 - m, d1 = x1 - m;
  float rs = rsqrtf(wave_sum(d0 * d0 + d1 * d1) * (1.0f / 64.0f) + 1e-6f);
  cx1[(long)row * C1_ + lane]      = (bf16_t)(d0 * rs * g1[lane] + b1[lane]);
  cx1[(long)row * C1_ + lane + 32] = (bf16_t)(d1 * rs * g1[lane + 32] + b1[lane + 32]);

  float e[6];
#pragma unroll
  for (int j = 0; j < 6; j++) {
    int c = lane + 32 * j;
    e[j] = (c < C1_) ? r1[c] : r2[c - C1_];
  }
  float s = 0.f;
#pragma unroll
  for (int j = 0; j < 6; j++) s += e[j];
  float ma = wave_sum(s) * (1.0f / 192.0f);
  float sq = 0.f;
#pragma unroll
  for (int j = 0; j < 6; j++) { e[j] -= ma; sq += e[j] * e[j]; }
  float rsa = rsqrtf(wave_sum(sq) * (1.0f / 192.0f) + 1e-6f);
#pragma unroll
  for (int j = 0; j < 6; j++) {
    int c = lane + 32 * j;
    ea[(long)row * KV_ + c] = (bf16_t)(e[j] * rsa * ga[c] + ba[c]);
  }
}

// ---------------------------------------------------------------------------
// LN over 64 channels (f32 in, bf16 out). 1 wave per row.
// ---------------------------------------------------------------------------
__global__ __launch_bounds__(256) void ln64_kernel(
    const float* __restrict__ x, const float* __restrict__ g,
    const float* __restrict__ b, bf16_t* __restrict__ y) {
  const int row  = blockIdx.x * 8 + (threadIdx.x >> 5);
  const int lane = threadIdx.x & 31;
  const float* r = x + (long)row * C1_;
  float x0 = r[lane], x1 = r[lane + 32];
  float m  = wave_sum(x0 + x1) * (1.0f / 64.0f);
  float d0 = x0 - m, d1 = x1 - m;
  float rs = rsqrtf(wave_sum(d0 * d0 + d1 * d1) * (1.0f / 64.0f) + 1e-6f);
  y[(long)row * C1_ + lane]      = (bf16_t)(d0 * rs * g[lane] + b[lane]);
  y[(long)row * C1_ + lane + 32] = (bf16_t)(d1 * rs * g[lane + 32] + b[lane + 32]);
}

// ---------------------------------------------------------------------------
// Generic GEMM: C[M,N] = A[M,K] * Bw[N,K]^T  (nn.Linear), bf16 WMMA, f32 acc.
// Tile: M=128, N=64, Kchunk=32, double-buffered async LDS staging.
// Epilogue compiled in via template params. TRANS stores C^T with a packed
// 16B store per accumulator (rows are contiguous in the [ch][m] layout).
// ---------------------------------------------------------------------------
template <bool BIAS, bool GELU, bool RESID, bool OUTF32, bool TRANS>
__global__ __launch_bounds__(256) void gemm_nt_kernel(
    const bf16_t* __restrict__ A, int lda, long aBatch,
    const bf16_t* __restrict__ Bw, int ldb, long bBatch,
    bf16_t* __restrict__ Cb, float* __restrict__ Cf, int ldc, long cBatch,
    int Kdim, const float* __restrict__ bias, const float* __restrict__ resid) {
  __shared__ bf16_t As[2][128][32];
  __shared__ bf16_t Bs[2][64][32];
  const int tid = threadIdx.x, lane = tid & 31, wave = tid >> 5, hi = lane >> 4;
  const long z = blockIdx.z;
  const bf16_t* Ab = A + z * aBatch;
  const bf16_t* Bb = Bw + z * bBatch;
  const int m0 = blockIdx.x * 128, n0 = blockIdx.y * 64;
  v8f acc[4] = {};
  const int ar = tid >> 2, ac = (tid & 3) * 8;
  const bf16_t* gA0 = &Ab[(long)(m0 + ar) * lda + ac];
  const bf16_t* gA1 = &Ab[(long)(m0 + ar + 64) * lda + ac];
  const bf16_t* gB  = &Bb[(long)(n0 + ar) * ldb + ac];

  auto stage = [&](int buf, int k0) {
    async_copy16(&As[buf][ar][ac],      gA0 + k0);
    async_copy16(&As[buf][ar + 64][ac], gA1 + k0);
    async_copy16(&Bs[buf][ar][ac],      gB  + k0);
  };

  const int nk = Kdim >> 5;
  stage(0, 0);
  for (int i = 0; i < nk; i++) {
    async_wait0();
    __syncthreads();
    if (i + 1 < nk) stage((i + 1) & 1, (i + 1) << 5);
    const int bu = i & 1;
    const int mrow = wave * 16 + (lane & 15);
    v16bf a;
#pragma unroll
    for (int e = 0; e < 16; e++) a[e] = As[bu][mrow][a_kidx(e, hi)];
#pragma unroll
    for (int nt = 0; nt < 4; nt++) {
      const int col = nt * 16 + (lane & 15);
      v16bf bb;
#pragma unroll
      for (int e = 0; e < 16; e++) bb[e] = Bs[bu][col][a_kidx(e, hi)];
      acc[nt] = __builtin_amdgcn_wmma_f32_16x16x32_bf16(
          false, a, false, bb, (short)0, acc[nt], false, false);
    }
  }

  const long cbase = z * cBatch;
#pragma unroll
  for (int nt = 0; nt < 4; nt++) {
    const int gn = n0 + nt * 16 + (lane & 15);
    float bs = 0.f;
    if (BIAS) bs = bias[gn];
    if (TRANS) {
      v8bf pk;
#pragma unroll
      for (int v = 0; v < 8; v++) pk[v] = (bf16_t)(acc[nt][v] + bs);
      const long gmb = m0 + wave * 16 + (hi ? 8 : 0);
      *(v8bf*)&Cb[cbase + (long)gn * ldc + gmb] = pk;   // 8 consecutive rows
    } else {
#pragma unroll
      for (int v = 0; v < 8; v++) {
        const int gm = m0 + wave * 16 + v + (hi ? 8 : 0);
        float x = acc[nt][v] + bs;
        if (GELU) x = 0.5f * x * (1.0f + erff(x * 0.70710678118f));
        if (RESID) x += resid[(long)gm * ldc + gn];
        if (OUTF32) Cf[cbase + (long)gm * ldc + gn] = x;
        else        Cb[cbase + (long)gm * ldc + gn] = (bf16_t)x;
      }
    }
  }
}

// ---------------------------------------------------------------------------
// Fused scores: S = Q^T K / sqrt(KV), InstanceNorm over 64x192, softmax over
// KV, store P row-major (bf16 [64][192]). One workgroup per (b,h).
// A = QT [H][64][B*N], B = KT [H][192][B*N]; contraction n, 98 pipelined
// chunks of 32 with async double-buffered LDS staging. All fragment reads are
// contiguous 16B ds_load_b128 pairs.
// ---------------------------------------------------------------------------
__global__ __launch_bounds__(256) void scores_softmax_kernel(
    const bf16_t* __restrict__ QT, const bf16_t* __restrict__ KT,
    bf16_t* __restrict__ P) {
  __shared__ bf16_t Aq[2][64][32];
  __shared__ bf16_t Bs[2][192][32];
  __shared__ float  Sb[64][196];
  __shared__ float  red[256];
  const int tid = threadIdx.x, lane = tid & 31, wave = tid >> 5, hi = lane >> 4;
  const int bh = blockIdx.x, b = bh >> 2, h = bh & 3;
  const bf16_t* Ab = QT + (long)h * C1_ * BN_ + (long)b * N_;  // row c, ld=BN_
  const bf16_t* Bb = KT + (long)h * KV_ * BN_ + (long)b * N_;  // row k, ld=BN_
  const int mt = wave & 3, ng = wave >> 2;
  v8f acc[6] = {};
  const int ar = tid >> 2, ac = (tid & 3) * 8;

  auto stage = [&](int buf, int n0) {
    async_copy16(&Aq[buf][ar][ac], &Ab[(long)ar * BN_ + n0 + ac]);
#pragma unroll
    for (int i = 0; i < 3; i++)
      async_copy16(&Bs[buf][ar + 64 * i][ac],
                   &Bb[(long)(ar + 64 * i) * BN_ + n0 + ac]);
  };

  const int nk = N_ / 32;
  stage(0, 0);
  for (int it = 0; it < nk; it++) {
    async_wait0();
    __syncthreads();
    if (it + 1 < nk) stage((it + 1) & 1, (it + 1) * 32);
    const int bu = it & 1;
    const int mrow = mt * 16 + (lane & 15);
    v16bf a;
#pragma unroll
    for (int e = 0; e < 16; e++) a[e] = Aq[bu][mrow][a_kidx(e, hi)];
#pragma unroll
    for (int t = 0; t < 6; t++) {
      const int col = (ng * 6 + t) * 16 + (lane & 15);
      v16bf bb;
#pragma unroll
      for (int e = 0; e < 16; e++) bb[e] = Bs[bu][col][a_kidx(e, hi)];
      acc[t] = __builtin_amdgcn_wmma_f32_16x16x32_bf16(
          false, a, false, bb, (short)0, acc[t], false, false);
    }
  }

  __syncthreads();
  const float sc = 0.072168783649f;  // 1/sqrt(192)
#pragma unroll
  for (int t = 0; t < 6; t++) {
    const int gn = (ng * 6 + t) * 16 + (lane & 15);
#pragma unroll
    for (int v = 0; v < 8; v++)
      Sb[mt * 16 + v + (hi ? 8 : 0)][gn] = acc[t][v] * sc;
  }
  __syncthreads();

  // InstanceNorm statistics over all 64*192 values (4 threads per row)
  float s = 0.f, sq = 0.f;
  {
    const int r0 = tid >> 2;
    const int c0 = (tid & 3) * 48;
    for (int j = 0; j < 48; j++) {
      float x = Sb[r0][c0 + j];
      s += x; sq += x * x;
    }
  }
  red[tid] = s; __syncthreads();
  for (int st = 128; st > 0; st >>= 1) { if (tid < st) red[tid] += red[tid + st]; __syncthreads(); }
  const float mean = red[0] * (1.0f / 12288.0f);
  __syncthreads();
  red[tid] = sq; __syncthreads();
  for (int st = 128; st > 0; st >>= 1) { if (tid < st) red[tid] += red[tid + st]; __syncthreads(); }
  const float var  = red[0] * (1.0f / 12288.0f) - mean * mean;
  const float rstd = rsqrtf(var + 1e-5f);
  __syncthreads();

  // Softmax over KV per row c; one wave per row, 8 rows per wave.
  // Output row-major P[c][k] -> contiguous stores.
  bf16_t* Pb = P + (long)bh * C1_ * KV_;
  for (int rr = 0; rr < 8; rr++) {
    const int r = wave * 8 + rr;
    float xs[6];
    float mx = -3.0e38f;
#pragma unroll
    for (int j = 0; j < 6; j++) {
      xs[j] = (Sb[r][lane + 32 * j] - mean) * rstd;
      mx = fmaxf(mx, xs[j]);
    }
    mx = wave_max(mx);
    float es = 0.f;
#pragma unroll
    for (int j = 0; j < 6; j++) { xs[j] = __expf(xs[j] - mx); es += xs[j]; }
    es = wave_sum(es);
    const float inv = 1.0f / es;
#pragma unroll
    for (int j = 0; j < 6; j++)
      Pb[(long)r * KV_ + lane + 32 * j] = (bf16_t)(xs[j] * inv);
  }
}

// ---------------------------------------------------------------------------
// ctx^T mean over heads: ctxm[b,n,c] = 0.25 * sum_h sum_k V[h,bn,k]*P[bh,c,k]
// Tile: 64 rows (n) x 64 cols (c). grid = (N/64, 1, B). 24 pipelined steps.
// B operand P is [64 c][192 k] row-major -> staged [col][k], contiguous reads.
// ---------------------------------------------------------------------------
__global__ __launch_bounds__(256) void ctx_mean_kernel(
    const bf16_t* __restrict__ V, const bf16_t* __restrict__ P,
    bf16_t* __restrict__ ctxm) {
  __shared__ bf16_t Vs[2][64][32];
  __shared__ bf16_t Ps[2][64][32];
  const int tid = threadIdx.x, lane = tid & 31, wave = tid >> 5, hi = lane >> 4;
  const int b = blockIdx.z;
  const int n0 = blockIdx.x * 64;
  const int mt = wave & 3, ng = wave >> 2;
  v8f acc[2] = {};
  const int ar = tid >> 2, ac = (tid & 3) * 8;

  auto stage = [&](int buf, int s) {
    const int h = s / 6, k0 = (s % 6) * 32;
    const bf16_t* Vb = V + ((long)h * BN_ + (long)b * N_ + n0) * KV_;
    const bf16_t* Pb = P + (long)(b * 4 + h) * C1_ * KV_;
    async_copy16(&Vs[buf][ar][ac], &Vb[(long)ar * KV_ + k0 + ac]);
    async_copy16(&Ps[buf][ar][ac], &Pb[(long)ar * KV_ + k0 + ac]);
  };

  const int nsteps = H_ * (KV_ / 32);  // 24
  stage(0, 0);
  for (int it = 0; it < nsteps; it++) {
    async_wait0();
    __syncthreads();
    if (it + 1 < nsteps) stage((it + 1) & 1, it + 1);
    const int bu = it & 1;
    const int mrow = mt * 16 + (lane & 15);
    v16bf a;
#pragma unroll
    for (int e = 0; e < 16; e++) a[e] = Vs[bu][mrow][a_kidx(e, hi)];
#pragma unroll
    for (int t = 0; t < 2; t++) {
      const int col = (ng * 2 + t) * 16 + (lane & 15);
      v16bf bb;
#pragma unroll
      for (int e = 0; e < 16; e++) bb[e] = Ps[bu][col][a_kidx(e, hi)];
      acc[t] = __builtin_amdgcn_wmma_f32_16x16x32_bf16(
          false, a, false, bb, (short)0, acc[t], false, false);
    }
  }

#pragma unroll
  for (int t = 0; t < 2; t++) {
    const int gn = (ng * 2 + t) * 16 + (lane & 15);
#pragma unroll
    for (int v = 0; v < 8; v++) {
      const int n = n0 + mt * 16 + v + (hi ? 8 : 0);
      ctxm[((long)b * N_ + n) * C1_ + gn] = (bf16_t)(acc[t][v] * 0.25f);
    }
  }
}

// ---------------------------------------------------------------------------
// Host-side orchestration
// ---------------------------------------------------------------------------
extern "C" void kernel_launch(void* const* d_in, const int* in_sizes, int n_in,
                              void* d_out, int out_size, void* d_ws, size_t ws_size,
                              hipStream_t stream) {
  const float* emb1    = (const float*)d_in[0];
  const float* emb2    = (const float*)d_in[1];
  const float* Wq      = (const float*)d_in[2];
  const float* Wk      = (const float*)d_in[3];
  const float* Wv      = (const float*)d_in[4];
  const float* Wout    = (const float*)d_in[5];
  const float* ln1_g   = (const float*)d_in[6];
  const float* ln1_b   = (const float*)d_in[7];
  const float* lnall_g = (const float*)d_in[8];
  const float* lnall_b = (const float*)d_in[9];
  const float* ffn_g   = (const float*)d_in[10];
  const float* ffn_b   = (const float*)d_in[11];
  const float* fc1_w   = (const float*)d_in[12];
  const float* fc1_b   = (const float*)d_in[13];
  const float* fc2_w   = (const float*)d_in[14];
  const float* fc2_b   = (const float*)d_in[15];
  float* out = (float*)d_out;
  (void)in_sizes; (void)n_in; (void)out_size; (void)ws_size;

  char* ws = (char*)d_ws;
  size_t off = 0;
  auto take = [&](size_t bytes) -> void* {
    off = (off + 255) & ~(size_t)255;
    void* p = ws + off;
    off += bytes;
    return p;
  };
  bf16_t* cx1   = (bf16_t*)take((size_t)BN_ * C1_ * 2);
  bf16_t* ea    = (bf16_t*)take((size_t)BN_ * KV_ * 2);
  bf16_t* Wqb   = (bf16_t*)take((size_t)H_ * C1_ * C1_ * 2);
  bf16_t* Wkb   = (bf16_t*)take((size_t)H_ * KV_ * KV_ * 2);
  bf16_t* Wvb   = (bf16_t*)take((size_t)H_ * KV_ * KV_ * 2);
  bf16_t* Woutb = (bf16_t*)take((size_t)C1_ * C1_ * 2);
  bf16_t* fc1b  = (bf16_t*)take((size_t)MLP_ * C1_ * 2);
  bf16_t* fc2b  = (bf16_t*)take((size_t)C1_ * MLP_ * 2);
  bf16_t* QT    = (bf16_t*)take((size_t)H_ * C1_ * BN_ * 2);   // [H][64][B*N]
  bf16_t* KT    = (bf16_t*)take((size_t)H_ * KV_ * BN_ * 2);   // [H][192][B*N]
  bf16_t* Vbuf  = (bf16_t*)take((size_t)H_ * BN_ * KV_ * 2);   // [H][B*N][192]
  bf16_t* Pm    = (bf16_t*)take((size_t)B_ * H_ * C1_ * KV_ * 2);
  bf16_t* ctxm  = (bf16_t*)take((size_t)BN_ * C1_ * 2);
  float*  cx    = (float*) take((size_t)BN_ * C1_ * 4);
  bf16_t* ybuf  = (bf16_t*)take((size_t)BN_ * C1_ * 2);
  bf16_t* h1    = (bf16_t*)take((size_t)BN_ * MLP_ * 2);

  auto cvt = [&](const float* s, bf16_t* d, int n) {
    f32_to_bf16_kernel<<<(n + 255) / 256, 256, 0, stream>>>(s, d, n);
  };
  cvt(Wq,    Wqb,   H_ * C1_ * C1_);
  cvt(Wk,    Wkb,   H_ * KV_ * KV_);
  cvt(Wv,    Wvb,   H_ * KV_ * KV_);
  cvt(Wout,  Woutb, C1_ * C1_);
  cvt(fc1_w, fc1b,  MLP_ * C1_);
  cvt(fc2_w, fc2b,  C1_ * MLP_);

  ln_concat_kernel<<<BN_ / 8, 256, 0, stream>>>(emb1, emb2, ln1_g, ln1_b,
                                                lnall_g, lnall_b, cx1, ea);

  // Q = cx1 * Wq[h]^T, stored transposed -> QT [H][64][B*N]
  gemm_nt_kernel<false, false, false, false, true>
      <<<dim3(BN_ / 128, 1, H_), 256, 0, stream>>>(
      cx1, C1_, 0L, Wqb, C1_, (long)C1_ * C1_,
      QT, nullptr, BN_, (long)C1_ * BN_, C1_, nullptr, nullptr);
  // K = ea * Wk[h]^T, stored transposed -> KT [H][192][B*N]
  gemm_nt_kernel<false, false, false, false, true>
      <<<dim3(BN_ / 128, KV_ / 64, H_), 256, 0, stream>>>(
      ea, KV_, 0L, Wkb, KV_, (long)KV_ * KV_,
      KT, nullptr, BN_, (long)KV_ * BN_, KV_, nullptr, nullptr);
  // V = ea * Wv[h]^T, row-major [H][B*N][192]
  gemm_nt_kernel<false, false, false, false, false>
      <<<dim3(BN_ / 128, KV_ / 64, H_), 256, 0, stream>>>(
      ea, KV_, 0L, Wvb, KV_, (long)KV_ * KV_,
      Vbuf, nullptr, KV_, (long)BN_ * KV_, KV_, nullptr, nullptr);

  // S = Q^T K / sqrt(KV); InstanceNorm; softmax; -> P [bh][64][192]
  scores_softmax_kernel<<<B_ * H_, 256, 0, stream>>>(QT, KT, Pm);

  // ctx^T mean over heads
  ctx_mean_kernel<<<dim3(N_ / 64, 1, B_), 256, 0, stream>>>(Vbuf, Pm, ctxm);

  // cx = ctxm * Wout^T + emb1   (f32)
  gemm_nt_kernel<false, false, true, true, false>
      <<<dim3(BN_ / 128, 1, 1), 256, 0, stream>>>(
      ctxm, C1_, 0L, Woutb, C1_, 0L,
      nullptr, cx, C1_, 0L, C1_, nullptr, emb1);

  ln64_kernel<<<BN_ / 8, 256, 0, stream>>>(cx, ffn_g, ffn_b, ybuf);

  // h1 = GELU(y * fc1^T + b1)   (bf16)
  gemm_nt_kernel<true, true, false, false, false>
      <<<dim3(BN_ / 128, MLP_ / 64, 1), 256, 0, stream>>>(
      ybuf, C1_, 0L, fc1b, C1_, 0L,
      h1, nullptr, MLP_, 0L, C1_, fc1_b, nullptr);

  // out = h1 * fc2^T + b2 + cx  (f32)
  gemm_nt_kernel<true, false, true, true, false>
      <<<dim3(BN_ / 128, 1, 1), 256, 0, stream>>>(
      h1, MLP_, 0L, fc2b, MLP_, 0L,
      nullptr, out, C1_, 0L, MLP_, fc2_b, cx);
}
